// TemporalLSTM_77455440216810
// MI455X (gfx1250) — matrix operand order repaired and checked by reading, not compiled, not run on hardware
//
#include <hip/hip_runtime.h>
#include <hip/hip_bf16.h>
#include <math.h>

#define INPUT_DIM 128
#define HIDDEN    256
#define SEQ       512
#define NOUT      24
#define TEMP      64
#define NB        512

typedef __attribute__((ext_vector_type(16))) __bf16        v16bf;
typedef __attribute__((ext_vector_type(8)))  float         v8f;
typedef __attribute__((ext_vector_type(4)))  unsigned int  v4u;

union Frag16 { v4u q[2]; unsigned int u[8]; v16bf v; };

// XOR swizzle on dword index at 8-dword granularity. Fragment j-runs start at
// multiples of 4 (A) / 8 (B) and the XOR only touches bits [5:3], so
// (d0 + j) ^ X == (d0 ^ X) + j  -- we apply the XOR to the run base ourselves
// so the backend sees manifestly consecutive addresses (-> ds_load_b128).
#define SWZ_BITS(r) (((r) & 7) << 3)

__device__ __forceinline__ float sigmoidf_(float x) {
    return 1.0f / (1.0f + expf(-x));
}

__device__ __forceinline__ unsigned short bf16_bits(__bf16 v) {
    return __builtin_bit_cast(unsigned short, v);
}

// ---------------------------------------------------------------------------
// init: h0 = embedding[:, :256], c0 = embedding[:, 256:], y_prev = y_0
// ---------------------------------------------------------------------------
__global__ __launch_bounds__(256) void init_kernel(
    const float* __restrict__ embedding, const float* __restrict__ y0,
    float* __restrict__ h, float* __restrict__ c, float* __restrict__ y_prev)
{
    int idx = blockIdx.x * 256 + threadIdx.x;     // 0 .. B*256-1
    int b = idx >> 8, j = idx & 255;
    h[idx] = embedding[b * (2 * HIDDEN) + j];
    c[idx] = embedding[b * (2 * HIDDEN) + HIDDEN + j];
    if (j == 0) y_prev[b] = y0[b];
}

// ---------------------------------------------------------------------------
// a = c_t @ Ua_t   : (B,256)x(256,64) -> (B,64)
// ---------------------------------------------------------------------------
__global__ __launch_bounds__(256) void a_kernel(
    const float* __restrict__ c, const float* __restrict__ Ua,
    float* __restrict__ a, int t)
{
    int idx = blockIdx.x * 256 + threadIdx.x;     // 0 .. B*64-1
    int b = idx >> 6, n = idx & 63;
    const float* cb = c + b * HIDDEN;
    const float* u  = Ua + t * TEMP + n;          // stride NOUT*TEMP per k
    float s = 0.0f;
    #pragma unroll 4
    for (int k = 0; k < HIDDEN; ++k)
        s += cb[k] * u[k * (NOUT * TEMP)];
    a[idx] = s;
}

// ---------------------------------------------------------------------------
// Attention scores: e[b,l] = Va_t . tanh( H[b,l,:] @ Wa_t + a[b,:] + ba_t )
// bf16x3 split-precision WMMA:  D ~= Ahi*Bhi + Ahi*Blo + Alo*Bhi, fp32 accum.
// Block = 128 threads (4 waves); each wave owns one 16-row l tile; block
// covers 64 rows of one b.  grid = dim3(SEQ/64, B).
// LDS: 4 bf16 planes, XOR-swizzled, 16 KB each = 64 KB total.
// ---------------------------------------------------------------------------
__global__ __launch_bounds__(128) void attn_kernel(
    const float* __restrict__ H,  const float* __restrict__ Wa,
    const float* __restrict__ ba, const float* __restrict__ Va,
    const float* __restrict__ a,  float* __restrict__ e, int t)
{
    __shared__ unsigned int sWaHi[TEMP * 64];       // Wa_t^T hi: [n][d], d=k/2
    __shared__ unsigned int sWaLo[TEMP * 64];       // Wa_t^T lo
    __shared__ unsigned int sHHi[4][16 * 64];       // per-wave H tile hi: [m][d]
    __shared__ unsigned int sHLo[4][16 * 64];       // per-wave H tile lo

    const int b   = blockIdx.y;
    const int tid = threadIdx.x;

    // Stage Wa_t (128x64 fp32) -> split bf16 hi/lo, transposed [n][k].
    // k-major iteration keeps the global reads coalesced; b16 LDS stores.
    for (int i = tid; i < INPUT_DIM * TEMP; i += 128) {
        int k = i >> 6, n = i & 63;
        float  f  = Wa[k * (NOUT * TEMP) + t * TEMP + n];
        __bf16 hi = (__bf16)f;
        __bf16 lo = (__bf16)(f - (float)hi);        // exact residual, then RNE
        int di = (k >> 1) ^ SWZ_BITS(n);
        ((__bf16*)sWaHi)[(n * 64 + di) * 2 + (k & 1)] = hi;
        ((__bf16*)sWaLo)[(n * 64 + di) * 2 + (k & 1)] = lo;
    }

    const int wv   = tid >> 5;
    const int lane = tid & 31;
    const int l0   = blockIdx.x * 64 + wv * 16;
    const float* Hb = H + ((size_t)b * SEQ + l0) * INPUT_DIM;
    unsigned int* hHi = &sHHi[wv][0];
    unsigned int* hLo = &sHLo[wv][0];

    // Stage this wave's 16x128 H tile -> split bf16 hi/lo.
    // One dword (two adjacent k) per lane per iteration: global b64 loads,
    // packed b32 LDS stores.
    for (int i = lane; i < 16 * 64; i += 32) {
        int m = i >> 6, d = i & 63;
        float  f0  = Hb[m * INPUT_DIM + 2 * d];
        float  f1  = Hb[m * INPUT_DIM + 2 * d + 1];
        __bf16 h0 = (__bf16)f0, h1 = (__bf16)f1;
        __bf16 l0v = (__bf16)(f0 - (float)h0);
        __bf16 l1v = (__bf16)(f1 - (float)h1);
        int idx = m * 64 + (d ^ SWZ_BITS(m));
        hHi[idx] = (unsigned int)bf16_bits(h0) | ((unsigned int)bf16_bits(h1) << 16);
        hLo[idx] = (unsigned int)bf16_bits(l0v) | ((unsigned int)bf16_bits(l1v) << 16);
    }
    __syncthreads();

    const int g   = lane >> 4;    // half-wave group
    const int nl  = lane & 15;    // A: row m ; B/D: col within 16-tile
    const int swz = SWZ_BITS(nl); // (n & 7) == (nl & 7) for all B rows used

    const v8f vzero = {0.f,0.f,0.f,0.f,0.f,0.f,0.f,0.f};
    v8f acc[4];
    #pragma unroll
    for (int nt = 0; nt < 4; ++nt) acc[nt] = vzero;

    // K = 128 = 4 tiles of 32 ; N = 64 = 4 tiles of 16 ; 3 WMMA terms each.
    #pragma unroll
    for (int kt = 0; kt < 4; ++kt) {
        const int kb2 = kt * 16;                    // kb / 2 (dwords)
        // A fragment: two aligned 4-dword runs per plane.
        // A-layout: V_j -> d = kb2 + 8g + j (matches B)
        Frag16 aHi, aLo;
        {
            const int d0 = kb2 + 8 * g;
            const int b0 = nl * 64 + (d0 ^ swz);
            const int b1 = nl * 64 + ((d0 + 4) ^ swz);
            aHi.q[0] = *(const v4u*)&hHi[b0];
            aHi.q[1] = *(const v4u*)&hHi[b1];
            aLo.q[0] = *(const v4u*)&hLo[b0];
            aLo.q[1] = *(const v4u*)&hLo[b1];
        }
        #pragma unroll
        for (int nt = 0; nt < 4; ++nt) {
            // B fragment: one aligned 8-dword run per plane.
            // B-layout (contiguous k, as A): V_j -> d = kb2 + 8g + j
            const int n  = nt * 16 + nl;
            const int bb = n * 64 + ((kb2 + 8 * g) ^ swz);
            Frag16 bHi, bLo;
            bHi.q[0] = *(const v4u*)&sWaHi[bb];
            bHi.q[1] = *(const v4u*)&sWaHi[bb + 4];
            bLo.q[0] = *(const v4u*)&sWaLo[bb];
            bLo.q[1] = *(const v4u*)&sWaLo[bb + 4];
            acc[nt] = __builtin_amdgcn_wmma_f32_16x16x32_bf16(
                false, aLo.v, false, bHi.v, (short)0, acc[nt], false, false);
            acc[nt] = __builtin_amdgcn_wmma_f32_16x16x32_bf16(
                false, aHi.v, false, bLo.v, (short)0, acc[nt], false, false);
            acc[nt] = __builtin_amdgcn_wmma_f32_16x16x32_bf16(
                false, aHi.v, false, bHi.v, (short)0, acc[nt], false, false);
        }
    }

    // Epilogue: e[b, l0+m] = sum_n Va[n] * tanh(D[m,n] + a[b,n] + ba[n])
    float partial[8];
    #pragma unroll
    for (int r = 0; r < 8; ++r) partial[r] = 0.0f;

    #pragma unroll
    for (int nt = 0; nt < 4; ++nt) {
        int n = nt * 16 + nl;
        float bias_n = a[b * TEMP + n] + ba[t * TEMP + n];
        float va_n   = Va[n * NOUT + t];
        #pragma unroll
        for (int r = 0; r < 8; ++r)
            partial[r] += va_n * tanhf(acc[nt][r] + bias_n);
    }
    // reduce over the 16 lanes of each half-wave (columns)
    #pragma unroll
    for (int off = 1; off < 16; off <<= 1) {
        #pragma unroll
        for (int r = 0; r < 8; ++r)
            partial[r] += __shfl_xor(partial[r], off, 32);
    }
    if (nl == 0) {
        #pragma unroll
        for (int r = 0; r < 8; ++r)
            e[(size_t)b * SEQ + l0 + r + 8 * g] = partial[r];   // row m = r + 8g
    }
}

// ---------------------------------------------------------------------------
// softmax over SEQ (with 1/sqrt(HS) scale) + ctx[b,d] = sum_l beta[l]*H[b,l,d]
// ---------------------------------------------------------------------------
__global__ __launch_bounds__(128) void softmax_ctx_kernel(
    const float* __restrict__ e, const float* __restrict__ H,
    float* __restrict__ ctx)
{
    __shared__ float sE[SEQ];
    __shared__ float red[128];
    const int b = blockIdx.x, tid = threadIdx.x;
    const float scale = 0.0625f;                  // 1/sqrt(256)

    float lm = -1e30f;
    for (int i = tid; i < SEQ; i += 128) {
        float v = e[(size_t)b * SEQ + i] * scale;
        sE[i] = v;
        lm = fmaxf(lm, v);
    }
    red[tid] = lm;
    __syncthreads();
    for (int s = 64; s > 0; s >>= 1) {
        if (tid < s) red[tid] = fmaxf(red[tid], red[tid + s]);
        __syncthreads();
    }
    float M = red[0];
    __syncthreads();

    float ls = 0.0f;
    for (int i = tid; i < SEQ; i += 128) {
        float v = expf(sE[i] - M);
        sE[i] = v;
        ls += v;
    }
    red[tid] = ls;
    __syncthreads();
    for (int s = 64; s > 0; s >>= 1) {
        if (tid < s) red[tid] += red[tid + s];
        __syncthreads();
    }
    float inv = 1.0f / red[0];
    __syncthreads();

    // ctx: thread d accumulates over l (coalesced across d each iteration)
    const float* Hb = H + (size_t)b * SEQ * INPUT_DIM;
    float acc = 0.0f;
    #pragma unroll 4
    for (int l = 0; l < SEQ; ++l)
        acc += sE[l] * Hb[(size_t)l * INPUT_DIM + tid];
    ctx[b * INPUT_DIM + tid] = acc * inv;
}

// ---------------------------------------------------------------------------
// gates = ctx@W_t + h@U_t + b_t + y_prev*Wy_t ; LSTM pointwise; y = h.fcw+fcb
// ---------------------------------------------------------------------------
__global__ __launch_bounds__(256) void gates_lstm_kernel(
    const float* __restrict__ ctx, const float* __restrict__ W,
    const float* __restrict__ U,   const float* __restrict__ bias,
    const float* __restrict__ Wy,  const float* __restrict__ fcw,
    const float* __restrict__ fcb, float* __restrict__ h,
    float* __restrict__ c, float* __restrict__ y_prev,
    float* __restrict__ out_y, float* __restrict__ out_h, int t)
{
    __shared__ float sc[INPUT_DIM];
    __shared__ float sh[HIDDEN];
    __shared__ float red[256];

    const int b = blockIdx.x, tid = threadIdx.x;
    if (tid < INPUT_DIM) sc[tid] = ctx[b * INPUT_DIM + tid];
    sh[tid] = h[b * HIDDEN + tid];
    const float yp = y_prev[b];
    __syncthreads();

    const float* Bt  = bias + t * (4 * HIDDEN);
    const float* Wyt = Wy   + t * (4 * HIDDEN);
    float acc[4];
    #pragma unroll
    for (int q = 0; q < 4; ++q)
        acc[q] = Bt[tid + 256 * q] + yp * Wyt[tid + 256 * q];

    #pragma unroll 2
    for (int k = 0; k < INPUT_DIM; ++k) {
        const float* wr = W + (size_t)k * (NOUT * 4 * HIDDEN) + t * (4 * HIDDEN);
        float cv = sc[k];
        #pragma unroll
        for (int q = 0; q < 4; ++q) acc[q] += cv * wr[tid + 256 * q];
    }
    #pragma unroll 2
    for (int k = 0; k < HIDDEN; ++k) {
        const float* ur = U + (size_t)k * (NOUT * 4 * HIDDEN) + t * (4 * HIDDEN);
        float hv = sh[k];
        #pragma unroll
        for (int q = 0; q < 4; ++q) acc[q] += hv * ur[tid + 256 * q];
    }

    float ig = sigmoidf_(acc[0]);
    float fg = sigmoidf_(acc[1]);
    float gg = tanhf(acc[2]);
    float og = sigmoidf_(acc[3]);
    float cn = fg * c[b * HIDDEN + tid] + ig * gg;
    float hn = og * tanhf(cn);
    c[b * HIDDEN + tid] = cn;
    h[b * HIDDEN + tid] = hn;
    out_h[((size_t)b * NOUT + t) * HIDDEN + tid] = hn;

    red[tid] = hn * fcw[t * HIDDEN + tid];
    __syncthreads();
    for (int s = 128; s > 0; s >>= 1) {
        if (tid < s) red[tid] += red[tid + s];
        __syncthreads();
    }
    if (tid == 0) {
        float y = red[0] + fcb[t];
        y_prev[b] = y;
        out_y[b * NOUT + t] = y;
    }
}

// ---------------------------------------------------------------------------
extern "C" void kernel_launch(void* const* d_in, const int* in_sizes, int n_in,
                              void* d_out, int out_size, void* d_ws, size_t ws_size,
                              hipStream_t stream)
{
    (void)in_sizes; (void)n_in; (void)out_size; (void)ws_size;
    const float* H    = (const float*)d_in[0];
    const float* y0   = (const float*)d_in[1];
    const float* emb  = (const float*)d_in[2];
    const float* Wa   = (const float*)d_in[3];
    const float* Ua   = (const float*)d_in[4];
    const float* ba   = (const float*)d_in[5];
    const float* Va   = (const float*)d_in[6];
    const float* W    = (const float*)d_in[7];
    const float* U    = (const float*)d_in[8];
    const float* bias = (const float*)d_in[9];
    const float* Wy   = (const float*)d_in[10];
    const float* fcw  = (const float*)d_in[11];
    const float* fcb  = (const float*)d_in[12];

    float* out   = (float*)d_out;
    float* out_y = out;                      // (B, NOUT)
    float* out_h = out + (size_t)NB * NOUT;  // (B, NOUT, HIDDEN)

    char* ws = (char*)d_ws;
    size_t off = 0;
    float* e      = (float*)(ws + off); off += (size_t)NB * SEQ * 4;        // 1 MB
    float* a      = (float*)(ws + off); off += (size_t)NB * TEMP * 4;       // 128 KB
    float* ctx    = (float*)(ws + off); off += (size_t)NB * INPUT_DIM * 4;  // 256 KB
    float* h      = (float*)(ws + off); off += (size_t)NB * HIDDEN * 4;     // 512 KB
    float* c      = (float*)(ws + off); off += (size_t)NB * HIDDEN * 4;     // 512 KB
    float* y_prev = (float*)(ws + off); off += (size_t)NB * 4;              // 2 KB

    init_kernel<<<NB, 256, 0, stream>>>(emb, y0, h, c, y_prev);

    for (int t = 0; t < NOUT; ++t) {
        a_kernel<<<(NB * TEMP) / 256, 256, 0, stream>>>(c, Ua, a, t);
        attn_kernel<<<dim3(SEQ / 64, NB), 128, 0, stream>>>(H, Wa, ba, Va, a, e, t);
        softmax_ctx_kernel<<<NB, 128, 0, stream>>>(e, H, ctx);
        gates_lstm_kernel<<<NB, 256, 0, stream>>>(ctx, W, U, bias, Wy, fcw, fcb,
                                                  h, c, y_prev, out_y, out_h, t);
    }
}
